// Transformer_4879082848941
// MI455X (gfx1250) — matrix operand order repaired and checked by reading, not compiled, not run on hardware
//
#include <hip/hip_runtime.h>
#include <math.h>

// ---------------- problem constants ----------------
constexpr int kDim    = 1024;
constexpr int kHeads  = 16;
constexpr int kDHead  = 64;
constexpr int kSeq    = 1024;
constexpr int kBatch  = 4;
constexpr int kTok    = kBatch * kSeq;      // 4096
constexpr int kDepth  = 6;
constexpr int kFFH    = 5460;               // 2*FF_INNER
constexpr int kFFHPad = 5504;               // mult of 128
constexpr int kFFI    = 2730;               // FF_INNER
constexpr int kFFIPad = 2752;               // mult of 32 (K pad)

// GEMM tiling
constexpr int kBM = 128;
constexpr int kBN = 128;
constexpr int kBK = 32;

typedef __attribute__((ext_vector_type(16))) __bf16 v16bf;
typedef __attribute__((ext_vector_type(8)))  float  v8f;
typedef unsigned int v4u __attribute__((ext_vector_type(4)));
typedef unsigned int v8u __attribute__((ext_vector_type(8)));

union FragBF { v16bf v; uint4 q[2]; };

__device__ __forceinline__ unsigned short f2bf(float f) {
  unsigned int u = __float_as_uint(f);
  u += 0x7fffu + ((u >> 16) & 1u);       // round-to-nearest-even
  return (unsigned short)(u >> 16);
}

// ---- CDNA5 async global->LDS copy (ASYNCcnt) --------------------------------
// Per-lane 16B copy, bypasses VGPRs. LDS operand = low 32 bits of the flat
// shared address (LDS aperture keeps the offset in addr[31:0]).
__device__ __forceinline__ void async_copy_b128(unsigned short* lds_dst,
                                                const unsigned short* gsrc) {
  const unsigned int l = (unsigned int)(size_t)lds_dst;
  asm volatile("global_load_async_to_lds_b128 %0, %1, off"
               :: "v"(l), "v"(gsrc) : "memory");
}
__device__ __forceinline__ void wait_async(int pending) {
  if (pending == 0)
    asm volatile("s_wait_asynccnt 0x0" ::: "memory");
  else
    asm volatile("s_wait_asynccnt 0x2" ::: "memory");
}

// ---- CDNA5 Tensor Data Mover: 2D tile (128 rows x 32 bf16) -> LDS -----------
// D# group0: count=1, lds_addr, 57-bit global_addr, type=2.
// D# group1: data_size=2B, tensor 32x128, tile 32x128, dim0 stride = K elems.
__device__ __forceinline__ void tdm_load_tile_b(unsigned short* lds_dst,
                                                const unsigned short* gsrc,
                                                int strideElems) {
  v4u g0;
  const unsigned long long ga = (unsigned long long)(size_t)gsrc;
  g0[0] = 1u;                                            // count=1, user mode
  g0[1] = (unsigned int)(size_t)lds_dst;                 // lds_addr (bytes)
  g0[2] = (unsigned int)(ga & 0xffffffffull);            // global_addr[31:0]
  g0[3] = (unsigned int)((ga >> 32) & 0x01ffffffull)     // global_addr[56:32]
          | (2u << 30);                                  // type = 2 ("image")
  v8u g1;
  g1[0] = 1u << 16;                                      // data_size = 1 -> 2B
  g1[1] = (32u & 0xffffu) << 16;                         // tensor_dim0 = 32
  g1[2] = (128u << 16);                                  // tensor_dim1 = 128
  g1[3] = (32u << 16);                                   // tile_dim0 = 32
  g1[4] = 128u;                                          // tile_dim1=128, tile_dim2=0
  g1[5] = (unsigned int)strideElems;                     // dim0 stride lo
  g1[6] = 0u;                                            // stride hi / dim1 stride lo
  g1[7] = 0u;
  asm volatile("tensor_load_to_lds %0, %1" :: "s"(g0), "s"(g1) : "memory");
}

// A-operand fragment (16x32 bf16): lanes 0-15 hold row M=lane, K = {0..7,16..23};
// lanes 16-31 hold same rows, K = {8..15,24..31}. rowptr -> element K=0 of the row.
__device__ __forceinline__ v16bf frag_a(const unsigned short* rowptr, int lane) {
  FragBF f;
  const int half = (lane & 16) ? 8 : 0;
  f.q[0] = *(const uint4*)(rowptr + half);
  f.q[1] = *(const uint4*)(rowptr + half + 16);
  return f.v;
}
// B-operand fragment (32x16 bf16, stored as B^T row = column N): lanes 0-15 hold
// N=lane, K = {0..7,16..23}; lanes 16-31 hold K = {8..15,24..31}. colptr -> element K=0.
__device__ __forceinline__ v16bf frag_b(const unsigned short* colptr, int lane) {
  FragBF f;
  const int half = (lane & 16) ? 8 : 0;
  f.q[0] = *(const uint4*)(colptr + half);
  f.q[1] = *(const uint4*)(colptr + half + 16);
  return f.v;
}

__device__ __forceinline__ v8f wmma_bf16(v16bf a, v16bf b, v8f c) {
  return __builtin_amdgcn_wmma_f32_16x16x32_bf16(false, a, false, b, (short)0, c,
                                                 false, false);
}

// ---------------- weight transpose-pack: W[K][N] f32 -> Bt[Npad][Kpad] bf16 ----
__global__ void pack_bt_kernel(const float* __restrict__ W,
                               unsigned short* __restrict__ Bt,
                               int K, int N, int Kpad, long long total) {
  const long long idx = (long long)blockIdx.x * blockDim.x + threadIdx.x;
  if (idx >= total) return;
  const int n = (int)(idx / Kpad);
  const int k = (int)(idx % Kpad);
  unsigned short v = 0;
  if (k < K && n < N) v = f2bf(W[(size_t)k * N + n]);
  Bt[idx] = v;
}

// ---------------- RMSNorm ----------------
__global__ __launch_bounds__(256) void rmsnorm_bf_kernel(
    const float* __restrict__ x, const float* __restrict__ gamma,
    unsigned short* __restrict__ out) {
  __shared__ float red[256];
  const int tok = blockIdx.x, tid = threadIdx.x;
  const float* xr = x + (size_t)tok * kDim;
  float s = 0.f;
  for (int j = tid; j < kDim; j += 256) { const float v = xr[j]; s += v * v; }
  red[tid] = s; __syncthreads();
  for (int st = 128; st > 0; st >>= 1) {
    if (tid < st) red[tid] += red[tid + st];
    __syncthreads();
  }
  const float n = fmaxf(sqrtf(red[0]), 1e-12f);
  const float r = 32.0f / n;            // sqrt(1024)/norm
  unsigned short* orow = out + (size_t)tok * kDim;
  for (int j = tid; j < kDim; j += 256) orow[j] = f2bf(xr[j] * r * gamma[j]);
}

__global__ __launch_bounds__(256) void rmsnorm_f32_kernel(
    const float* __restrict__ x, const float* __restrict__ gamma,
    float* __restrict__ out) {
  __shared__ float red[256];
  const int tok = blockIdx.x, tid = threadIdx.x;
  const float* xr = x + (size_t)tok * kDim;
  float s = 0.f;
  for (int j = tid; j < kDim; j += 256) { const float v = xr[j]; s += v * v; }
  red[tid] = s; __syncthreads();
  for (int st = 128; st > 0; st >>= 1) {
    if (tid < st) red[tid] += red[tid + st];
    __syncthreads();
  }
  const float n = fmaxf(sqrtf(red[0]), 1e-12f);
  const float r = 32.0f / n;
  float* orow = out + (size_t)tok * kDim;
  for (int j = tid; j < kDim; j += 256) orow[j] = xr[j] * r * gamma[j];
}

// ---------------- T5 relative-position bias table: [H][S][S] ----------------
__global__ void bias_kernel(const float* __restrict__ rel_emb,
                            float* __restrict__ bias) {
  const int idx = blockIdx.x * blockDim.x + threadIdx.x;
  if (idx >= kSeq * kSeq) return;
  const int i = idx / kSeq, j = idx % kSeq;
  int nn = i - j; if (nn < 0) nn = 0;
  int bucket;
  if (nn < 16) {
    bucket = nn;
  } else {
    bucket = 16 + (int)(logf((float)nn * (1.0f / 16.0f)) * (16.0f / logf(8.0f)));
    if (bucket > 31) bucket = 31;
  }
  for (int h = 0; h < kHeads; h++)
    bias[((size_t)h * kSeq + i) * kSeq + j] = rel_emb[bucket * kHeads + h] * 8.0f;
}

// ---------------- WMMA GEMM: C[M][ldc] (+=) alpha*A[M][K]·B + bias ------------
// A bf16 row-major [M][K]; Bt bf16 = B^T row-major [Ngrid][K].
// 256 threads = 8 waves (2x4); wave tile 64x32 = 4x2 WMMA accumulators.
// Double-buffered LDS: A via async global->LDS copies (ASYNCcnt), B via a
// single TDM descriptor per K-step issued by wave 0 (TENSORcnt).
__global__ __launch_bounds__(256) void gemm_bf16_kernel(
    const unsigned short* __restrict__ A,
    const unsigned short* __restrict__ Bt,
    float* __restrict__ C,
    unsigned short* __restrict__ Cbf,      // optional bf16 mirror of C
    const float* __restrict__ bias,        // optional [Nc]
    int K, int ldc, int Nc, float alpha, int accum) {
  __shared__ unsigned short As[2][kBM][kBK];
  __shared__ unsigned short Bs[2][kBN][kBK];

  const int tid  = threadIdx.x;
  const int lane = tid & 31;
  const int wave = tid >> 5;
  const int wm   = wave >> 2;     // 0..1 -> 64-row slab
  const int wn   = wave & 3;      // 0..3 -> 32-col slab
  const int m0   = blockIdx.y * kBM;
  const int n0   = blockIdx.x * kBN;

  v8f acc[4][2];
  for (int mt = 0; mt < 4; mt++)
    for (int nt = 0; nt < 2; nt++)
      for (int i = 0; i < 8; i++) acc[mt][nt][i] = 0.f;

  const int lrow = tid >> 1;             // 0..127
  const int lcol = (tid & 1) * 16;       // 0 or 16
  const unsigned short* arow = A + (size_t)(m0 + lrow) * K + lcol;

  // prologue: stage tile 0 into buffer 0
  async_copy_b128(&As[0][lrow][lcol], arow);
  async_copy_b128(&As[0][lrow][lcol + 8], arow + 8);
  if (wave == 0) tdm_load_tile_b(&Bs[0][0][0], Bt + (size_t)n0 * K, K);

  const int KT = K / kBK;
  for (int kt = 0; kt < KT; ++kt) {
    const int cur = kt & 1;
    const int nxt = cur ^ 1;
    if (kt + 1 < KT) {                   // software pipeline: stage tile kt+1
      const int k1 = (kt + 1) * kBK;
      async_copy_b128(&As[nxt][lrow][lcol], arow + k1);
      async_copy_b128(&As[nxt][lrow][lcol + 8], arow + k1 + 8);
      if (wave == 0) tdm_load_tile_b(&Bs[nxt][0][0], Bt + (size_t)n0 * K + k1, K);
      wait_async(2);                     // tile kt landed; kt+1 still in flight
      if (wave == 0) __builtin_amdgcn_s_wait_tensorcnt(1);
    } else {
      wait_async(0);
      if (wave == 0) __builtin_amdgcn_s_wait_tensorcnt(0);
    }
    __syncthreads();                     // all waves' tile-kt data visible

    v16bf af[4];
    for (int mt = 0; mt < 4; mt++)
      af[mt] = frag_a(&As[cur][wm * 64 + mt * 16 + (lane & 15)][0], lane);
    v16bf bf2[2];
    for (int nt = 0; nt < 2; nt++)
      bf2[nt] = frag_b(&Bs[cur][wn * 32 + nt * 16 + (lane & 15)][0], lane);
    for (int mt = 0; mt < 4; mt++)
      for (int nt = 0; nt < 2; nt++)
        acc[mt][nt] = wmma_bf16(af[mt], bf2[nt], acc[mt][nt]);

    __syncthreads();                     // buffer reusable for tile kt+2
  }

  const int nl = lane & 15;
  const int mb = (lane >> 4) * 8;
  for (int mt = 0; mt < 4; mt++) {
    for (int nt = 0; nt < 2; nt++) {
      const int gn = n0 + wn * 32 + nt * 16 + nl;
      if (gn >= Nc) continue;
      const float bv = bias ? bias[gn] : 0.f;
      for (int i = 0; i < 8; i++) {
        const int gm = m0 + wm * 64 + mt * 16 + mb + i;
        float v = alpha * acc[mt][nt][i] + bv;
        float* cp = C + (size_t)gm * ldc + gn;
        if (accum) v += *cp;
        *cp = v;
        if (Cbf) Cbf[(size_t)gm * ldc + gn] = f2bf(v);
      }
    }
  }
}

// ---------------- KV split: kv f32 [TOK][2048] -> K bf16, V^T bf16 ------------
__global__ void kvpost_kernel(const float* __restrict__ kv,
                              unsigned short* __restrict__ Kb,
                              unsigned short* __restrict__ Vt) {
  const long long idx = (long long)blockIdx.x * blockDim.x + threadIdx.x;
  if (idx >= (long long)kTok * 2048) return;
  const int tok = (int)(idx / 2048);
  const int c   = (int)(idx % 2048);
  const float v = kv[idx];
  if (c < kDim) {
    Kb[(size_t)tok * kDim + c] = f2bf(v);
  } else {
    const int cc = c - kDim, h = cc / kDHead, d = cc % kDHead;
    const int b = tok / kSeq, n = tok % kSeq;
    Vt[(size_t)((b * kHeads + h) * kDHead + d) * kSeq + n] = f2bf(v);
  }
}

// ---------------- flash attention per (b, h, 64-query tile) -------------------
__global__ __launch_bounds__(128) void attention_kernel(
    const unsigned short* __restrict__ Q,    // [TOK][DIM], pre-scaled
    const unsigned short* __restrict__ Kb,   // [TOK][DIM]
    const unsigned short* __restrict__ Vt,   // [B][H][DHEAD][SEQ]
    const float* __restrict__ bias,          // [H][SEQ][SEQ]
    unsigned short* __restrict__ O) {        // [TOK][DIM]
  __shared__ unsigned short Qs[64][64];
  __shared__ unsigned short Ks[64][64];
  __shared__ unsigned short Vs[64][64];
  __shared__ unsigned short Ps[4][16][64];

  const int tid  = threadIdx.x;
  const int lane = tid & 31;
  const int wave = tid >> 5;     // wave owns 16 query rows
  const int qt = blockIdx.x, h = blockIdx.y, b = blockIdx.z;
  const int q0 = qt * 64;
  const int nl = lane & 15;
  const int mb = (lane >> 4) * 8;

  // Q tile via async global->LDS
  for (int c = tid; c < 512; c += 128) {
    const int row = c >> 3, kc = (c & 7) * 8;
    async_copy_b128(&Qs[row][kc],
                    Q + (size_t)(b * kSeq + q0 + row) * kDim + h * kDHead + kc);
  }
  wait_async(0);
  __syncthreads();

  float mrow[8], lrow[8];
  v8f oacc[4];
  for (int i = 0; i < 8; i++) { mrow[i] = -3.0e38f; lrow[i] = 0.f; }
  for (int dt = 0; dt < 4; dt++)
    for (int i = 0; i < 8; i++) oacc[dt][i] = 0.f;

  for (int kt = 0; kt <= qt; kt++) {       // causal: skip fully-masked tiles
    const int k0 = kt * 64;
    __syncthreads();                       // done reading previous K/V tile
    for (int c = tid; c < 512; c += 128) {
      const int row = c >> 3, kc = (c & 7) * 8;
      async_copy_b128(&Ks[row][kc],
                      Kb + (size_t)(b * kSeq + k0 + row) * kDim + h * kDHead + kc);
      async_copy_b128(&Vs[row][kc],
                      Vt + (size_t)((b * kHeads + h) * kDHead + row) * kSeq + k0 + kc);
    }
    wait_async(0);
    __syncthreads();

    // S = Q·K^T  (16 q-rows x 64 keys per wave)
    v8f sacc[4];
    for (int nt = 0; nt < 4; nt++)
      for (int i = 0; i < 8; i++) sacc[nt][i] = 0.f;
    for (int ks = 0; ks < 2; ks++) {
      v16bf aq = frag_a(&Qs[wave * 16 + nl][ks * 32], lane);
      for (int nt = 0; nt < 4; nt++) {
        v16bf bk = frag_b(&Ks[nt * 16 + nl][ks * 32], lane);
        sacc[nt] = wmma_bf16(aq, bk, sacc[nt]);
      }
    }

    // bias + causal mask + online softmax
    float rmax[8];
    for (int i = 0; i < 8; i++) rmax[i] = -3.0e38f;
    for (int nt = 0; nt < 4; nt++) {
      for (int i = 0; i < 8; i++) {
        const int qrow = q0 + wave * 16 + mb + i;
        const int kcol = k0 + nt * 16 + nl;
        float s = sacc[nt][i] + bias[((size_t)h * kSeq + qrow) * kSeq + kcol];
        if (kcol > qrow) s = -1.0e30f;
        sacc[nt][i] = s;
        rmax[i] = fmaxf(rmax[i], s);
      }
    }
    for (int i = 0; i < 8; i++)
      for (int off = 8; off >= 1; off >>= 1)
        rmax[i] = fmaxf(rmax[i], __shfl_xor(rmax[i], off, 32));

    float scale[8], rsum[8];
    for (int i = 0; i < 8; i++) {
      const float mnew = fmaxf(mrow[i], rmax[i]);
      scale[i] = __expf(mrow[i] - mnew);
      mrow[i] = mnew;
      rsum[i] = 0.f;
    }
    for (int nt = 0; nt < 4; nt++) {
      for (int i = 0; i < 8; i++) {
        const float p = __expf(sacc[nt][i] - mrow[i]);
        sacc[nt][i] = p;
        rsum[i] += p;
      }
    }
    for (int i = 0; i < 8; i++)
      for (int off = 8; off >= 1; off >>= 1)
        rsum[i] += __shfl_xor(rsum[i], off, 32);
    for (int i = 0; i < 8; i++) lrow[i] = lrow[i] * scale[i] + rsum[i];
    for (int dt = 0; dt < 4; dt++)
      for (int i = 0; i < 8; i++) oacc[dt][i] *= scale[i];

    // P -> LDS (C-layout -> A-layout re-fragmentation)
    for (int nt = 0; nt < 4; nt++)
      for (int i = 0; i < 8; i++)
        Ps[wave][mb + i][nt * 16 + nl] = f2bf(sacc[nt][i]);
    __syncthreads();

    // O += P·V
    for (int ks = 0; ks < 2; ks++) {
      v16bf ap = frag_a(&Ps[wave][nl][ks * 32], lane);
      for (int dt = 0; dt < 4; dt++) {
        v16bf bv = frag_b(&Vs[dt * 16 + nl][ks * 32], lane);
        oacc[dt] = wmma_bf16(ap, bv, oacc[dt]);
      }
    }
  }

  for (int dt = 0; dt < 4; dt++) {
    for (int i = 0; i < 8; i++) {
      const int qrow = q0 + wave * 16 + mb + i;
      const float v = oacc[dt][i] / lrow[i];
      O[(size_t)(b * kSeq + qrow) * kDim + h * kDHead + dt * 16 + nl] = f2bf(v);
    }
  }
}

// ---------------- GEGLU gate: ffh f32 [TOK][5460] -> gated bf16 [TOK][2752] --
__global__ void gelugate_kernel(const float* __restrict__ hbuf,
                                unsigned short* __restrict__ out) {
  const long long idx = (long long)blockIdx.x * blockDim.x + threadIdx.x;
  if (idx >= (long long)kTok * kFFIPad) return;
  const int tok = (int)(idx / kFFIPad);
  const int c   = (int)(idx % kFFIPad);
  unsigned short r = 0;
  if (c < kFFI) {
    const float a = hbuf[(size_t)tok * kFFH + c];
    const float g = hbuf[(size_t)tok * kFFH + kFFI + c];
    const float gel = 0.5f * g * (1.0f + erff(g * 0.70710678118f));
    r = f2bf(gel * a);
  }
  out[idx] = r;
}

// ---------------- host orchestration ----------------
extern "C" void kernel_launch(void* const* d_in, const int* in_sizes, int n_in,
                              void* d_out, int out_size, void* d_ws, size_t ws_size,
                              hipStream_t stream) {
  (void)in_sizes; (void)n_in; (void)out_size; (void)ws_size;
  const float* x_in   = (const float*)d_in[0];
  const float* rel    = (const float*)d_in[1];
  const float* g_attn = (const float*)d_in[2];
  const float* wq     = (const float*)d_in[3];
  const float* wkv    = (const float*)d_in[4];
  const float* wo     = (const float*)d_in[5];
  const float* g_ff   = (const float*)d_in[6];
  const float* wff1   = (const float*)d_in[7];
  const float* bff1   = (const float*)d_in[8];
  const float* wff2   = (const float*)d_in[9];
  const float* bff2   = (const float*)d_in[10];
  const float* g_fin  = (const float*)d_in[11];
  float* out = (float*)d_out;

  char* ws = (char*)d_ws;
  size_t off = 0;
  auto alloc = [&](size_t bytes) -> char* {
    char* p = ws + off;
    off = (off + bytes + 255) & ~(size_t)255;
    return p;
  };
  float*  bias_t  = (float*)alloc((size_t)kHeads * kSeq * kSeq * 4);
  float*  x_cur   = (float*)alloc((size_t)kTok * kDim * 4);
  unsigned short* xn_bf   = (unsigned short*)alloc((size_t)kTok * kDim * 2);
  float*  cf      = (float*)alloc((size_t)kTok * kDim * 4);
  float*  kv_f    = (float*)alloc((size_t)kTok * 2048 * 4);
  float*  ffh_f   = (float*)alloc((size_t)kTok * kFFH * 4);
  unsigned short* q_bf    = (unsigned short*)alloc((size_t)kTok * kDim * 2);
  unsigned short* k_bf    = (unsigned short*)alloc((size_t)kTok * kDim * 2);
  unsigned short* vT_bf   = (unsigned short*)alloc((size_t)kTok * kDim * 2);
  unsigned short* attn_bf = (unsigned short*)alloc((size_t)kTok * kDim * 2);
  unsigned short* gate_bf = (unsigned short*)alloc((size_t)kTok * kFFIPad * 2);
  unsigned short* wqT     = (unsigned short*)alloc((size_t)kDim * kDim * 2);
  unsigned short* wkvT    = (unsigned short*)alloc((size_t)2048 * kDim * 2);
  unsigned short* woT     = (unsigned short*)alloc((size_t)kDim * kDim * 2);
  unsigned short* wff1T   = (unsigned short*)alloc((size_t)kFFHPad * kDim * 2);
  unsigned short* wff2T   = (unsigned short*)alloc((size_t)kDim * kFFIPad * 2);

  hipMemcpyAsync(x_cur, x_in, (size_t)kTok * kDim * 4,
                 hipMemcpyDeviceToDevice, stream);
  bias_kernel<<<(kSeq * kSeq + 255) / 256, 256, 0, stream>>>(rel, bias_t);

  for (int l = 0; l < kDepth; l++) {
    long long t;
    t = (long long)kDim * kDim;
    pack_bt_kernel<<<(unsigned)((t + 255) / 256), 256, 0, stream>>>(
        wq + (size_t)l * kDim * kDim, wqT, kDim, kDim, kDim, t);
    t = (long long)2048 * kDim;
    pack_bt_kernel<<<(unsigned)((t + 255) / 256), 256, 0, stream>>>(
        wkv + (size_t)l * kDim * 2048, wkvT, kDim, 2048, kDim, t);
    t = (long long)kDim * kDim;
    pack_bt_kernel<<<(unsigned)((t + 255) / 256), 256, 0, stream>>>(
        wo + (size_t)l * kDim * kDim, woT, kDim, kDim, kDim, t);
    t = (long long)kFFHPad * kDim;
    pack_bt_kernel<<<(unsigned)((t + 255) / 256), 256, 0, stream>>>(
        wff1 + (size_t)l * kDim * kFFH, wff1T, kDim, kFFH, kDim, t);
    t = (long long)kDim * kFFIPad;
    pack_bt_kernel<<<(unsigned)((t + 255) / 256), 256, 0, stream>>>(
        wff2 + (size_t)l * kFFI * kDim, wff2T, kFFI, kDim, kFFIPad, t);

    // ---- attention block ----
    rmsnorm_bf_kernel<<<kTok, 256, 0, stream>>>(x_cur, g_attn + l * kDim, xn_bf);
    gemm_bf16_kernel<<<dim3(kDim / kBN, kTok / kBM), 256, 0, stream>>>(
        xn_bf, wqT, cf, q_bf, nullptr, kDim, kDim, kDim, 0.125f, 0);  // q * d^-1/2
    gemm_bf16_kernel<<<dim3(2048 / kBN, kTok / kBM), 256, 0, stream>>>(
        xn_bf, wkvT, kv_f, nullptr, nullptr, kDim, 2048, 2048, 1.0f, 0);
    kvpost_kernel<<<(unsigned)(((long long)kTok * 2048 + 255) / 256), 256, 0, stream>>>(
        kv_f, k_bf, vT_bf);
    attention_kernel<<<dim3(kSeq / 64, kHeads, kBatch), 128, 0, stream>>>(
        q_bf, k_bf, vT_bf, bias_t, attn_bf);
    gemm_bf16_kernel<<<dim3(kDim / kBN, kTok / kBM), 256, 0, stream>>>(
        attn_bf, woT, x_cur, nullptr, nullptr, kDim, kDim, kDim, 1.0f, 1);  // +residual

    // ---- feedforward block ----
    rmsnorm_bf_kernel<<<kTok, 256, 0, stream>>>(x_cur, g_ff + l * kDim, xn_bf);
    gemm_bf16_kernel<<<dim3(kFFHPad / kBN, kTok / kBM), 256, 0, stream>>>(
        xn_bf, wff1T, ffh_f, nullptr, bff1 + (size_t)l * kFFH, kDim, kFFH, kFFH,
        1.0f, 0);
    gelugate_kernel<<<(unsigned)(((long long)kTok * kFFIPad + 255) / 256), 256, 0,
                      stream>>>(ffh_f, gate_bf);
    gemm_bf16_kernel<<<dim3(kDim / kBN, kTok / kBM), 256, 0, stream>>>(
        gate_bf, wff2T, x_cur, nullptr, bff2 + (size_t)l * kDim, kFFIPad, kDim,
        kDim, 1.0f, 1);  // +residual
  }

  rmsnorm_f32_kernel<<<kTok, 256, 0, stream>>>(x_cur, g_fin, out);
}
